// LSTM_1D_TimeSeries_77549929496949
// MI455X (gfx1250) — compile-verified, matrix operand order checked by reading, not run on hardware
//
#include <hip/hip_runtime.h>
#include <hip/hip_bf16.h>

typedef __attribute__((ext_vector_type(16))) __bf16 v16bf;
typedef __attribute__((ext_vector_type(8)))  float  v8f;
typedef __attribute__((ext_vector_type(4)))  float  v4f;

#define T_STEPS 512
#define BATCH   512
#define HDIM    256
#define BT      16    // batch rows per workgroup
#define NW      16    // waves per workgroup (512 threads)

__device__ __forceinline__ float fast_tanh(float x) {
#if __has_builtin(__builtin_amdgcn_tanhf)
    return __builtin_amdgcn_tanhf(x);
#else
    float e = __expf(2.0f * x);
    return (e - 1.0f) / (e + 1.0f);
#endif
}

// ---------------------------------------------------------------------------
// Kernel 1: one-time f32 -> bf16 conversion of W_hh into workspace
// ---------------------------------------------------------------------------
__global__ void convert_whh_kernel(const float* __restrict__ w,
                                   __bf16* __restrict__ o, int n) {
    int i = blockIdx.x * blockDim.x + threadIdx.x;
    if (i < n) o[i] = (__bf16)w[i];
}

// ---------------------------------------------------------------------------
// Kernel 2: persistent LSTM. One workgroup = 16 batch rows, runs all 512 steps.
// Wave w owns hidden columns [w*16, w*16+16). Weight placement (to fit the
// 256 directly-addressable VGPRs without spilling):
//   gates i,f (W_hh rows 0..511)    -> per-wave register slices (128 VGPRs)
//   gates g,o (W_hh rows 512..1023) -> bf16 copy in LDS (256 KB)
// ---------------------------------------------------------------------------
__global__ __launch_bounds__(NW * 32, 1)
void lstm_persist_kernel(const float* __restrict__ ts,     // [B, T]
                         const float* __restrict__ W_ih,   // [4H, 1]
                         const float* __restrict__ b_ih,   // [4H]
                         const float* __restrict__ b_hh,   // [4H]
                         const float* __restrict__ W_out,  // [1, T*H]
                         const float* __restrict__ b_out,  // [1]
                         const __bf16* __restrict__ Whh,   // [4H, H] bf16
                         float* __restrict__ out)          // [B, 1]
{
    extern __shared__ __attribute__((aligned(32))) char smem[];
    // [512 rows][HDIM] bf16 : W_hh rows 512..1023 (gates g,o)  : 256 KB
    __bf16 (*whh_lds)[HDIM] = (__bf16 (*)[HDIM])smem;
    // [2][BT][HDIM] bf16 double-buffered hidden state          :  16 KB
    __bf16 (*hbuf)[BT][HDIM] =
        (__bf16 (*)[BT][HDIM])(smem + 2 * HDIM * HDIM * 2);
    // [T][BT] f32 transposed time-series tile                  :  32 KB
    float (*ts_lds)[BT] =
        (float (*)[BT])(smem + 2 * HDIM * HDIM * 2 + 2 * BT * HDIM * 2);
    // [NW][BT] f32 per-wave output partials                    :   1 KB
    float (*out_red)[BT] =
        (float (*)[BT])(smem + 2 * HDIM * HDIM * 2 + 2 * BT * HDIM * 2 +
                        T_STEPS * BT * 4);

    const int tid   = threadIdx.x;
    const int wave  = tid >> 5;     // 0..15
    const int lane  = tid & 31;
    const int l16   = lane & 15;
    const int hl    = lane >> 4;    // 0 or 1 (lane half)
    const int bbase = blockIdx.x * BT;

    // ---- stage W_hh rows 512..1023 (gates g,o) into LDS ------------------
    // Prefer the CDNA5 async global->LDS DMA path (ASYNCcnt-tracked) when the
    // toolchain exposes it; fall back to a plain coalesced copy otherwise.
#if __has_builtin(__builtin_amdgcn_global_load_async_to_lds_b128) && \
    __has_builtin(__builtin_amdgcn_s_wait_asynccnt)
    {
        typedef int vint4 __attribute__((vector_size(16)));
        typedef vint4 __attribute__((address_space(1)))* g4p;
        typedef vint4 __attribute__((address_space(3)))* l4p;
        char* srcb = (char*)(Whh + (size_t)512 * HDIM);   // const dropped
        char* dstb = (char*)&whh_lds[0][0];
        for (int i = tid; i < (512 * HDIM * 2) / 16; i += NW * 32) {
            __builtin_amdgcn_global_load_async_to_lds_b128(
                (g4p)(srcb + 16 * i), (l4p)(dstb + 16 * i), 0, 0);
        }
        __builtin_amdgcn_s_wait_asynccnt(0);
    }
#else
    {
        const uint4* src = (const uint4*)(Whh + (size_t)512 * HDIM);
        uint4*       dst = (uint4*)&whh_lds[0][0];
        for (int i = tid; i < (512 * HDIM * 2) / 16; i += NW * 32)
            dst[i] = src[i];
    }
#endif
    // ---- stage ts tile (transposed) into LDS: ts_lds[t][b_local] ---------
    {
        int b  = tid >> 5;          // 0..15
        int t0 = (tid & 31) * 16;   // 16 consecutive t per thread
        const float* src = ts + (size_t)(bbase + b) * T_STEPS + t0;
        for (int i = 0; i < 16; ++i) ts_lds[t0 + i][b] = src[i];
    }
    // ---- zero initial hidden state (buffer 0) ----------------------------
    {
        __bf16* p = &hbuf[0][0][0];
        for (int i = tid; i < BT * HDIM; i += NW * 32) p[i] = (__bf16)0.0f;
    }

    // ---- per-lane scalar constants for this wave's gate columns ----------
    const int ncol = wave * 16 + l16;          // hidden column 0..255
    float wih_l[4], bias_l[4];
    #pragma unroll
    for (int g = 0; g < 4; ++g) {
        int r = g * HDIM + ncol;
        wih_l[g]  = W_ih[r];
        bias_l[g] = b_ih[r] + b_hh[r];
    }

    // ---- register-resident W_hh slices for gates i,f ---------------------
    // B-tile [K=32 x N=16] per (gate,kt): lane holds 16 contiguous bf16 of
    // row (g*256+ncol); K-half selected by lane half (matches A packing).
    v16bf wreg[2][8];
    #pragma unroll
    for (int g = 0; g < 2; ++g) {
        const __bf16* base = Whh + (size_t)(g * HDIM + ncol) * HDIM;
        #pragma unroll
        for (int kt = 0; kt < 8; ++kt)
            wreg[g][kt] = *(const v16bf*)(base + kt * 32 + hl * 16);
    }

    v8f   c_state = {};    // c for row m = hl*8 + v
    float oacc[8] = {};    // output-projection accumulators

    __syncthreads();

    int cur = 0;
    for (int t = 0; t < T_STEPS; ++t) {
        // -------- GEMM: gates = h @ Whh^T for this wave's 4x16 columns ----
        v8f acc[4] = {};
        #pragma unroll
        for (int kt = 0; kt < 8; ++kt) {
            const int ko = kt * 32 + hl * 16;
            v16bf a  = *(const v16bf*)(&hbuf[cur][l16][ko]);
            v16bf b2 = *(const v16bf*)(&whh_lds[ncol][ko]);         // gate g
            v16bf b3 = *(const v16bf*)(&whh_lds[HDIM + ncol][ko]);  // gate o
            acc[0] = __builtin_amdgcn_wmma_f32_16x16x32_bf16(
                false, a, false, wreg[0][kt], (short)0, acc[0], false, false);
            acc[1] = __builtin_amdgcn_wmma_f32_16x16x32_bf16(
                false, a, false, wreg[1][kt], (short)0, acc[1], false, false);
            acc[2] = __builtin_amdgcn_wmma_f32_16x16x32_bf16(
                false, a, false, b2, (short)0, acc[2], false, false);
            acc[3] = __builtin_amdgcn_wmma_f32_16x16x32_bf16(
                false, a, false, b3, (short)0, acc[3], false, false);
        }

        // output-projection weight for this lane at step t (L2 resident)
        float wov = W_out[t * HDIM + ncol];
        if (t + 1 < T_STEPS)
            __builtin_prefetch(&W_out[(t + 1) * HDIM + ncol], 0, 0);

        // -------- gate nonlinearities + state update ----------------------
        const int nxt = cur ^ 1;
        float tsv[8];
        *(v4f*)&tsv[0] = *(const v4f*)(&ts_lds[t][hl * 8]);
        *(v4f*)&tsv[4] = *(const v4f*)(&ts_lds[t][hl * 8 + 4]);
        #pragma unroll
        for (int v = 0; v < 8; ++v) {
            float tv = tsv[v];
            float gi = acc[0][v] + tv * wih_l[0] + bias_l[0];
            float gf = acc[1][v] + tv * wih_l[1] + bias_l[1];
            float gg = acc[2][v] + tv * wih_l[2] + bias_l[2];
            float go = acc[3][v] + tv * wih_l[3] + bias_l[3];
            float si = 1.0f / (1.0f + __expf(-gi));
            float sf = 1.0f / (1.0f + __expf(-gf));
            float so = 1.0f / (1.0f + __expf(-go));
            float cn = sf * c_state[v] + si * fast_tanh(gg);
            float hn = so * fast_tanh(cn);
            c_state[v] = cn;
            oacc[v] += hn * wov;
            hbuf[nxt][hl * 8 + v][ncol] = (__bf16)hn;
        }
        cur ^= 1;
        __syncthreads();
    }

    // -------- reduce output partials over this wave's 16 columns ----------
    #pragma unroll
    for (int v = 0; v < 8; ++v) {
        float s = oacc[v];
        s += __shfl_xor(s, 8, 32);
        s += __shfl_xor(s, 4, 32);
        s += __shfl_xor(s, 2, 32);
        s += __shfl_xor(s, 1, 32);
        oacc[v] = s;               // lane halves hold independent row sums
    }
    if (l16 == 0) {
        #pragma unroll
        for (int v = 0; v < 8; ++v)
            out_red[wave][hl * 8 + v] = oacc[v];
    }
    __syncthreads();

    // -------- deterministic cross-wave sum; each row written exactly once -
    if (tid < BT) {
        float s = b_out[0];
        #pragma unroll
        for (int w2 = 0; w2 < NW; ++w2) s += out_red[w2][tid];
        out[bbase + tid] = s;
    }
}

// ---------------------------------------------------------------------------
extern "C" void kernel_launch(void* const* d_in, const int* in_sizes, int n_in,
                              void* d_out, int out_size, void* d_ws, size_t ws_size,
                              hipStream_t stream) {
    const float* ts    = (const float*)d_in[0];
    const float* W_ih  = (const float*)d_in[1];
    const float* W_hh  = (const float*)d_in[2];
    const float* b_ih  = (const float*)d_in[3];
    const float* b_hh  = (const float*)d_in[4];
    const float* W_out = (const float*)d_in[5];
    const float* b_out = (const float*)d_in[6];
    float* out = (float*)d_out;

    __bf16* whh_bf = (__bf16*)d_ws;   // 4H*H bf16 = 512 KB

    const int nWhh = 4 * HDIM * HDIM;
    convert_whh_kernel<<<(nWhh + 255) / 256, 256, 0, stream>>>(W_hh, whh_bf, nWhh);

    size_t shmem = (size_t)(2 * HDIM * HDIM * 2)    // whh_lds  256 KB
                 + (size_t)(2 * BT * HDIM * 2)      // hbuf      16 KB
                 + (size_t)(T_STEPS * BT * 4)       // ts_lds    32 KB
                 + (size_t)(NW * BT * 4);           // out_red    1 KB
    lstm_persist_kernel<<<BATCH / BT, NW * 32, shmem, stream>>>(
        ts, W_ih, b_ih, b_hh, W_out, b_out, whh_bf, out);
}
